// MultiHeadAttention_49108656063284
// MI455X (gfx1250) — matrix operand order repaired and checked by reading, not compiled, not run on hardware
//
#include <hip/hip_runtime.h>

// ---------------------------------------------------------------------------
// MI455X (gfx1250) multi-head attention forward.
//   qkv = x @ Wqkv + bqkv      (bf16 WMMA, f32 accum)
//   attn = softmax(q k^T / 8)  (flash-style online softmax)
//   o    = attn @ v
//   out  = o @ Wout + bout     (f32 output)
// ---------------------------------------------------------------------------

typedef unsigned short u16;
typedef __attribute__((ext_vector_type(4)))  unsigned int uint32x4;
typedef __attribute__((ext_vector_type(4)))  float        float32x4;
typedef __attribute__((ext_vector_type(16))) __bf16      v16bf;
typedef __attribute__((ext_vector_type(8)))  float       v8f;

union FragU  { uint32x4 q[2]; v16bf v; };
union Pack16 { uint32x4 q[2]; u16 u[16]; };
union Pack32 { uint32x4 q[4]; u16 u[32]; };
union Pack8  { uint32x4 q;    u16 u[8];  };

__device__ __forceinline__ u16 f2bf(float f) {
  union { float f; unsigned u; } x; x.f = f;
  unsigned r = x.u + 0x7FFFu + ((x.u >> 16) & 1u);   // round-to-nearest-even
  return (u16)(r >> 16);
}

// CDNA5 WMMA 16x16x32 bf16 operand layouts (CDNA5 ISA §7.12.2).
// Element index (bf16 units) inside one 512-element (16x32) A tile:
//   lane = M + 16*((K&8)!=0), elem = (K&7) | ((K&16)>>1)
__device__ __forceinline__ int a_swz(int m, int k) {
  return (((k & 8) ? 16 : 0) + m) * 16 + ((k & 7) | ((k & 16) >> 1));
}
// One 512-element (32x16) B tile: lane = N + 16*((K&8)!=0), elem = (K&7) | ((K&16)>>1)
__device__ __forceinline__ int b_swz(int n, int k) {
  return (((k & 8) ? 16 : 0) + n) * 16 + ((k & 7) | ((k & 16) >> 1));
}

__device__ __forceinline__ v16bf load_frag(const u16* p) {
  FragU f;
  f.q[0] = *(const uint32x4*)(p);
  f.q[1] = *(const uint32x4*)(p + 8);
  return f.v;
}

__device__ __forceinline__ v8f zero8() {
  v8f z;
  #pragma unroll
  for (int i = 0; i < 8; ++i) z[i] = 0.0f;
  return z;
}

__device__ __forceinline__ v8f wmma_bf16(v16bf a, v16bf b, v8f c) {
  return __builtin_amdgcn_wmma_f32_16x16x32_bf16(false, a, false, b,
                                                 (short)0, c, false, false);
}

// ---------------------------------------------------------------------------
// fp32 -> bf16-bits conversion, 8 elements / thread, 128-bit memory ops
// ---------------------------------------------------------------------------
__global__ void cvt_f32_bf16(const float* __restrict__ in,
                             u16* __restrict__ out, int n8) {
  int i = blockIdx.x * blockDim.x + threadIdx.x;
  if (i >= n8) return;
  const float32x4* p = (const float32x4*)(in + (size_t)i * 8);
  float32x4 f0 = p[0], f1 = p[1];
  Pack8 o;
  #pragma unroll
  for (int j = 0; j < 4; ++j) { o.u[j] = f2bf(f0[j]); o.u[4 + j] = f2bf(f1[j]); }
  *(uint32x4*)(out + (size_t)i * 8) = o.q;
}

// ---------------------------------------------------------------------------
// GEMM: C[M,N] = A[M,K] * B[K,N] + bias[N]
// A,B bf16-bits row-major. Output bf16 or f32 (compile-time).
// Block tile 128x128, K-step 32, 256 threads (8 waves).
// Wave blocking: 32 rows x 64 cols (2 row-tiles x 4 col-tiles) -> per K-step
// 12 ds_load_b128 feed 8 WMMAs. Global loads register-double-buffered; LDS
// ping-pong -> single barrier per K-step.
// ---------------------------------------------------------------------------
template <bool BF16OUT>
__global__ __launch_bounds__(256) void gemm_bf16_wmma(
    const u16* __restrict__ A, const u16* __restrict__ B,
    const float* __restrict__ bias,
    u16* __restrict__ Cbf, float* __restrict__ Cf,
    int M, int N, int K)
{
  __shared__ u16 sA[2][8 * 512];   // 128x32 A block, 8 swizzled 16x32 tiles
  __shared__ u16 sB[2][8 * 512];   // 32x128 B block, 8 swizzled 32x16 tiles

  const int tid  = threadIdx.x;
  const int wave = tid >> 5;
  const int lane = tid & 31;
  const int mBase = blockIdx.y * 128;
  const int nBase = blockIdx.x * 128;
  const int wp = wave & 3;    // row pair: row-tiles {2wp, 2wp+1}
  const int wc = wave >> 2;   // col half: col-tiles {4wc .. 4wc+3}

  // staging assignment
  const int am = tid >> 1, ach = tid & 1;      // A: row am, k-chunk ach
  const int bk = tid >> 3, bc = tid & 7;       // B: row bk, n-chunk bc
  const u16* aSrc = A + (size_t)(mBase + am) * K + ach * 16;
  const u16* bSrc = B + (size_t)bk * N + nBase + bc * 16;
  const int at = am >> 4, amr = am & 15;

  Pack16 ra, rb;
  ra.q[0] = *(const uint32x4*)(aSrc);
  ra.q[1] = *(const uint32x4*)(aSrc + 8);
  rb.q[0] = *(const uint32x4*)(bSrc);
  rb.q[1] = *(const uint32x4*)(bSrc + 8);

  v8f acc[2][4];
  #pragma unroll
  for (int r = 0; r < 2; ++r)
    #pragma unroll
    for (int j = 0; j < 4; ++j) acc[r][j] = zero8();

  const int nsteps = K >> 5;
  for (int s = 0; s < nsteps; ++s) {
    const int buf = s & 1;
    // scatter prefetched registers into swizzled LDS
    #pragma unroll
    for (int i = 0; i < 16; ++i)
      sA[buf][at * 512 + a_swz(amr, ach * 16 + i)] = ra.u[i];
    #pragma unroll
    for (int i = 0; i < 16; ++i)
      sB[buf][bc * 512 + b_swz(i, bk)] = rb.u[i];
    __syncthreads();

    // issue global loads for next K-step (latency hidden under WMMAs)
    if (s + 1 < nsteps) {
      const u16* an = aSrc + (s + 1) * 32;
      const u16* bn = bSrc + (size_t)(s + 1) * 32 * N;
      ra.q[0] = *(const uint32x4*)(an);
      ra.q[1] = *(const uint32x4*)(an + 8);
      rb.q[0] = *(const uint32x4*)(bn);
      rb.q[1] = *(const uint32x4*)(bn + 8);
    }

    // compute: 2 A-frags x 4 B-frags
    v16bf a0 = load_frag(&sA[buf][(wp * 2 + 0) * 512 + lane * 16]);
    v16bf a1 = load_frag(&sA[buf][(wp * 2 + 1) * 512 + lane * 16]);
    #pragma unroll
    for (int j = 0; j < 4; ++j) {
      v16bf fb = load_frag(&sB[buf][(wc * 4 + j) * 512 + lane * 16]);
      acc[0][j] = wmma_bf16(a0, fb, acc[0][j]);
      acc[1][j] = wmma_bf16(a1, fb, acc[1][j]);
    }
    // no second barrier: next step's scatter targets the other LDS buffer,
    // and a wave can be at most one barrier ahead of any other wave.
  }

  // epilogue: C-frag layout -> lane gives N, vgpr gives M (+8 upper half)
  const int off8 = (lane >> 4) << 3;
  const int nl   = lane & 15;
  #pragma unroll
  for (int rt = 0; rt < 2; ++rt) {
    #pragma unroll
    for (int j = 0; j < 4; ++j) {
      int n = nBase + (wc * 4 + j) * 16 + nl;
      float bv = bias[n];
      #pragma unroll
      for (int v = 0; v < 8; ++v) {
        int m = mBase + (wp * 2 + rt) * 16 + v + off8;
        float val = acc[rt][j][v] + bv;
        if (BF16OUT) Cbf[(size_t)m * N + n] = f2bf(val);
        else         Cf [(size_t)m * N + n] = val;
      }
    }
  }
}

// ---------------------------------------------------------------------------
// Flash attention: block = 128 query rows of one (b,h); 8 waves * 16 rows.
// KV tiles of 64 keys; online softmax in f32; P and operands in bf16.
// Next tile's K/V global loads are register-prefetched during compute.
// qkv: [8192 x 3072] bf16-bits (q|k|v blocks of 1024 cols, head = 64 cols)
// out: [8192 x 1024] bf16-bits (row = b*2048+s, col = h*64+d)
// ---------------------------------------------------------------------------
#define SEQ 2048

__global__ __launch_bounds__(256) void attn_flash_wmma(
    const u16* __restrict__ qkv, u16* __restrict__ out)
{
  __shared__ u16 sQ[16 * 512];  // 8 waves x 2 kdim-steps, A-layout
  __shared__ u16 sK[ 8 * 512];  // (ks 0..1, keytile 0..3), B-layout (d x key)
  __shared__ u16 sV[ 8 * 512];  // (kvstep 0..1, dtile 0..3), B-layout (key x d)
  __shared__ u16 sP[16 * 512];  // per-wave P, A-layout

  const int tid  = threadIdx.x;
  const int wave = tid >> 5;
  const int lane = tid & 31;
  const int off8 = (lane >> 4) << 3;
  const int nl   = lane & 15;

  const int bh = blockIdx.y;
  const int b  = bh >> 4;
  const int h  = bh & 15;
  const int qm = b * SEQ + blockIdx.x * 128;     // global row base of Q block
  const int qc = h * 64, kc = 1024 + h * 64, vc = 2048 + h * 64;

  { // stage Q block (128 x 64) into A-swizzled LDS
    int r  = tid >> 1;
    int hf = tid & 1;                            // which 32-wide k-dim half
    const u16* src = qkv + (size_t)(qm + r) * 3072 + qc + hf * 32;
    Pack32 pv;
    pv.q[0] = *(const uint32x4*)(src);
    pv.q[1] = *(const uint32x4*)(src + 8);
    pv.q[2] = *(const uint32x4*)(src + 16);
    pv.q[3] = *(const uint32x4*)(src + 24);
    int mt = r >> 4, mr = r & 15;
    #pragma unroll
    for (int i = 0; i < 32; ++i)
      sQ[(mt * 2 + hf) * 512 + a_swz(mr, i)] = pv.u[i];
  }
  __syncthreads();
  const v16bf qa0 = load_frag(&sQ[(wave * 2 + 0) * 512 + lane * 16]);
  const v16bf qa1 = load_frag(&sQ[(wave * 2 + 1) * 512 + lane * 16]);

  // K/V staging assignment + register prefetch of tile 0
  const int key = tid >> 2, seg = tid & 3;
  const int ct  = key >> 4, krr = key & 15, ks = seg >> 1;  // for sK scatter
  const int kvs = key >> 5, dtl = seg;                      // for sV scatter
  const u16* kSrc = qkv + (size_t)(b * SEQ + key) * 3072 + kc + seg * 16;
  const u16* vSrc = qkv + (size_t)(b * SEQ + key) * 3072 + vc + seg * 16;
  const size_t tileStride = (size_t)64 * 3072;

  Pack16 rk, rv;
  rk.q[0] = *(const uint32x4*)(kSrc);
  rk.q[1] = *(const uint32x4*)(kSrc + 8);
  rv.q[0] = *(const uint32x4*)(vSrc);
  rv.q[1] = *(const uint32x4*)(vSrc + 8);

  float rm[8], rl[8];
  #pragma unroll
  for (int v = 0; v < 8; ++v) { rm[v] = -1e30f; rl[v] = 0.0f; }
  v8f ofr[4];
  #pragma unroll
  for (int dt = 0; dt < 4; ++dt) ofr[dt] = zero8();

  for (int kt = 0; kt < SEQ / 64; ++kt) {
    // scatter prefetched K tile: B-frags for Q*K^T (k-dim = d)
    #pragma unroll
    for (int i = 0; i < 16; ++i) {
      int d = seg * 16 + i;
      sK[(ks * 4 + ct) * 512 + b_swz(krr, d & 31)] = rk.u[i];
    }
    // scatter prefetched V tile: B-frags for P*V (k-dim = key)
    #pragma unroll
    for (int i = 0; i < 16; ++i)
      sV[(kvs * 4 + dtl) * 512 + b_swz(i, key & 31)] = rv.u[i];
    __syncthreads();

    // prefetch next tile's K/V while we compute
    if (kt + 1 < SEQ / 64) {
      const u16* kn = kSrc + (size_t)(kt + 1) * tileStride;
      const u16* vn = vSrc + (size_t)(kt + 1) * tileStride;
      rk.q[0] = *(const uint32x4*)(kn);
      rk.q[1] = *(const uint32x4*)(kn + 8);
      rv.q[0] = *(const uint32x4*)(vn);
      rv.q[1] = *(const uint32x4*)(vn + 8);
    }

    // S = (Q K^T) * 1/sqrt(64); 4 col-tiles of 16 keys each
    v8f sfr[4];
    #pragma unroll
    for (int c = 0; c < 4; ++c) {
      v8f s = zero8();
      s = wmma_bf16(qa0, load_frag(&sK[(0 * 4 + c) * 512 + lane * 16]), s);
      s = wmma_bf16(qa1, load_frag(&sK[(1 * 4 + c) * 512 + lane * 16]), s);
      sfr[c] = s * 0.125f;
    }

    // online softmax (rows live per-(vgpr, lane-half); reduce over 16 lanes)
    #pragma unroll
    for (int v = 0; v < 8; ++v) {
      float mx = rm[v];
      #pragma unroll
      for (int c = 0; c < 4; ++c) mx = fmaxf(mx, sfr[c][v]);
      #pragma unroll
      for (int d = 1; d < 16; d <<= 1) mx = fmaxf(mx, __shfl_xor(mx, d));
      float alpha = __expf(rm[v] - mx);
      rm[v] = mx;
      float sum = 0.0f;
      #pragma unroll
      for (int c = 0; c < 4; ++c) {
        float e = __expf(sfr[c][v] - mx);
        sfr[c][v] = e;                       // reuse sfr as P storage
        sum += e;
      }
      #pragma unroll
      for (int d = 1; d < 16; d <<= 1) sum += __shfl_xor(sum, d);
      rl[v] = rl[v] * alpha + sum;
      #pragma unroll
      for (int dt = 0; dt < 4; ++dt) ofr[dt][v] *= alpha;
    }

    // write P into wave-private A-layout LDS region (C-layout -> A-layout)
    #pragma unroll
    for (int c = 0; c < 4; ++c) {
      int col = c * 16 + nl;
      int pks = col >> 5, kin = col & 31;
      #pragma unroll
      for (int v = 0; v < 8; ++v)
        sP[(wave * 2 + pks) * 512 + a_swz(v + off8, kin)] = f2bf(sfr[c][v]);
    }

    // O += P * V
    v16bf pa0 = load_frag(&sP[(wave * 2 + 0) * 512 + lane * 16]);
    v16bf pa1 = load_frag(&sP[(wave * 2 + 1) * 512 + lane * 16]);
    #pragma unroll
    for (int dt = 0; dt < 4; ++dt) {
      ofr[dt] = wmma_bf16(pa0, load_frag(&sV[(0 * 4 + dt) * 512 + lane * 16]), ofr[dt]);
      ofr[dt] = wmma_bf16(pa1, load_frag(&sV[(1 * 4 + dt) * 512 + lane * 16]), ofr[dt]);
    }
    __syncthreads();   // protect sK/sV before next iteration's scatter
  }

  // normalize and store O as bf16 [8192 x 1024]
  #pragma unroll
  for (int v = 0; v < 8; ++v) {
    float inv = 1.0f / rl[v];
    int m = qm + wave * 16 + v + off8;
    #pragma unroll
    for (int dt = 0; dt < 4; ++dt) {
      int col = h * 64 + dt * 16 + nl;
      out[(size_t)m * 1024 + col] = f2bf(ofr[dt][v] * inv);
    }
  }
}

// ---------------------------------------------------------------------------
// launch
// ---------------------------------------------------------------------------
extern "C" void kernel_launch(void* const* d_in, const int* in_sizes, int n_in,
                              void* d_out, int out_size, void* d_ws, size_t ws_size,
                              hipStream_t stream)
{
  const float* x    = (const float*)d_in[0];   // [4,2048,1024]
  const float* Wqkv = (const float*)d_in[1];   // [1024,3072]
  const float* bqkv = (const float*)d_in[2];   // [3072]
  const float* Wout = (const float*)d_in[3];   // [1024,1024]
  const float* bout = (const float*)d_in[4];   // [1024]
  float* outp = (float*)d_out;                 // [4,2048,1024] f32

  char* ws = (char*)d_ws;
  u16* x_bf    = (u16*)(ws);                    // 8192*1024*2  = 16,777,216 B
  u16* wqkv_bf = (u16*)(ws + 16777216);         // 1024*3072*2  =  6,291,456 B
  u16* wout_bf = (u16*)(ws + 23068672);         // 1024*1024*2  =  2,097,152 B
  u16* qkv_bf  = (u16*)(ws + 25165824);         // 8192*3072*2  = 50,331,648 B
  u16* att_bf  = (u16*)(ws + 75497472);         // 8192*1024*2  = 16,777,216 B

  cvt_f32_bf16<<<(8388608 / 8 + 255) / 256, 256, 0, stream>>>(x, x_bf, 8388608 / 8);
  cvt_f32_bf16<<<(3145728 / 8 + 255) / 256, 256, 0, stream>>>(Wqkv, wqkv_bf, 3145728 / 8);
  cvt_f32_bf16<<<(1048576 / 8 + 255) / 256, 256, 0, stream>>>(Wout, wout_bf, 1048576 / 8);

  dim3 g1(3072 / 128, 8192 / 128);  // QKV projection -> bf16 qkv + bias
  gemm_bf16_wmma<true><<<g1, 256, 0, stream>>>(x_bf, wqkv_bf, bqkv,
                                               qkv_bf, nullptr, 8192, 3072, 1024);

  dim3 g2(SEQ / 128, 4 * 16);       // flash attention per (b,h)
  attn_flash_wmma<<<g2, 256, 0, stream>>>(qkv_bf, att_bf);

  dim3 g3(1024 / 128, 8192 / 128);  // output projection -> f32 + bias
  gemm_bf16_wmma<false><<<g3, 256, 0, stream>>>(att_bf, wout_bf, bout,
                                                nullptr, outp, 8192, 1024, 1024);

  (void)in_sizes; (void)n_in; (void)out_size; (void)ws_size;
}
